// ResLayer_85555748536423
// MI455X (gfx1250) — matrix operand order repaired and checked by reading, not compiled, not run on hardware
//
#include <hip/hip_runtime.h>

#define HEADS 6
#define CDIM 32
#define DIM 192
#define NEG_SLOPE 0.2f

typedef __attribute__((ext_vector_type(16))) __bf16 v16bf;
typedef __attribute__((ext_vector_type(8)))  float  v8f;

union FragBF16 { v16bf v; uint4 q[2]; };

// ---------------- precision conversion ----------------
__global__ void cvt_x_kernel(const float* __restrict__ x, __bf16* __restrict__ xb, int n) {
    int i = blockIdx.x * blockDim.x + threadIdx.x;
    if (i < n) xb[i] = (__bf16)x[i];
}

__global__ void cvt_w_kernel(const float* __restrict__ Wl, const float* __restrict__ Wr,
                             __bf16* __restrict__ Wb) {
    int i = blockIdx.x * blockDim.x + threadIdx.x;   // [0, 2*192*192)
    if (i >= 2 * DIM * DIM) return;
    int half = i / (DIM * DIM);
    int j = i - half * DIM * DIM;
    Wb[i] = (__bf16)(half ? Wr[j] : Wl[j]);
}

// ---------------- WMMA GEMM: Y[n, 0:384) = X @ [W_l; W_r]^T + bias ----------------
// One wave per 16x16 output tile; K-loop of 6 x (16x16x32 bf16 WMMA).
__global__ void gemm_wmma_kernel(const __bf16* __restrict__ xb, const __bf16* __restrict__ Wb,
                                 const float* __restrict__ bl, const float* __restrict__ br,
                                 float* __restrict__ xl, float* __restrict__ xr, int Nn) {
    const int wave = threadIdx.x >> 5;
    const int lane = threadIdx.x & 31;
    const int half = lane >> 4;       // K-half select (ISA 16-bit A/B layout)
    const int r    = lane & 15;
    const int m0 = blockIdx.x * 16;
    const int n0 = (blockIdx.y * 4 + wave) * 16;   // [0, 384)

    int arowi = m0 + r; if (arowi >= Nn) arowi = Nn - 1;
    // A (16x32 bf16): lane half=0 -> K = {k0..k0+7, k0+16..k0+23}; half=1 -> +8
    const __bf16* arow = xb + (size_t)arowi * DIM + 8 * half;
    // B (32x16 bf16): lane = column; half=0 -> K = {k0..k0+7, k0+16..k0+23}; half=1 -> +8
    const __bf16* brow = Wb + (size_t)(n0 + r) * DIM + 8 * half;

    v8f c = {};
#pragma unroll
    for (int k0 = 0; k0 < DIM; k0 += 32) {
        FragBF16 a, b;
        a.q[0] = *(const uint4*)(arow + k0);
        a.q[1] = *(const uint4*)(arow + k0 + 16);
        b.q[0] = *(const uint4*)(brow + k0);
        b.q[1] = *(const uint4*)(brow + k0 + 16);
        c = __builtin_amdgcn_wmma_f32_16x16x32_bf16(false, a.v, false, b.v,
                                                    (short)0, c, false, false);
    }

    // D layout: col = lane&15, row = m0 + i + 8*(lane>>4)
    const int col = n0 + r;
    float* Y; int cc; float bias;
    if (col < DIM) { Y = xl; cc = col;       bias = bl[cc]; }
    else           { Y = xr; cc = col - DIM; bias = br[cc]; }
    const int rowbase = m0 + 8 * half;
#pragma unroll
    for (int i = 0; i < 8; ++i) {
        int row = rowbase + i;
        if (row < Nn) Y[(size_t)row * DIM + cc] = c[i] + bias;
    }
}

// ---------------- segment softmax helpers ----------------
__device__ __forceinline__ int flipf(float f) {
    int i = __float_as_int(f);
    return i < 0 ? (i ^ 0x7fffffff) : i;          // monotone float->int map
}
__device__ __forceinline__ float unflipi(int i) {
    return __int_as_float(i < 0 ? (i ^ 0x7fffffff) : i);
}

__global__ void init_kernel(float* __restrict__ out, int* __restrict__ smax,
                            float* __restrict__ denom, int n_out, int n_nh) {
    int i = blockIdx.x * blockDim.x + threadIdx.x;
    if (i < n_out) out[i] = 0.0f;
    if (i < n_nh) { smax[i] = (int)0x80000000; denom[i] = 0.0f; }
}

// One wave32 per edge; lane = channel within head; 6 heads looped.
__global__ void edge_score_kernel(const int* __restrict__ ei, const float* __restrict__ xl,
                                  const float* __restrict__ xr, const float* __restrict__ att,
                                  float* __restrict__ score, int* __restrict__ smax,
                                  int E, int NE) {
    int e = blockIdx.x * (blockDim.x >> 5) + (threadIdx.x >> 5);
    if (e >= NE) return;
    const int lane = threadIdx.x & 31;
    int src, dst;
    if (e < E) { src = ei[e]; dst = ei[E + e]; } else { src = dst = e - E; }
    const float* ps = xl + (size_t)src * DIM;
    const float* pd = xr + (size_t)dst * DIM;
#pragma unroll
    for (int h = 0; h < HEADS; ++h) {
        float v = ps[h * CDIM + lane] + pd[h * CDIM + lane];
        v = v > 0.0f ? v : NEG_SLOPE * v;
        float p = v * att[h * CDIM + lane];
#pragma unroll
        for (int o = 16; o > 0; o >>= 1) p += __shfl_xor(p, o, 32);
        if (lane == 0) {
            score[(size_t)e * HEADS + h] = p;
            atomicMax(&smax[dst * HEADS + h], flipf(p));
        }
    }
}

__global__ void edge_exp_kernel(const int* __restrict__ ei, float* __restrict__ score,
                                const int* __restrict__ smax, float* __restrict__ denom,
                                int E, int NE) {
    int idx = blockIdx.x * blockDim.x + threadIdx.x;
    if (idx >= NE * HEADS) return;
    int e = idx / HEADS, h = idx - e * HEADS;
    int dst = (e < E) ? ei[E + e] : (e - E);
    float sm = unflipi(smax[dst * HEADS + h]);
    float ex = __expf(score[idx] - sm);
    score[idx] = ex;
    atomicAdd(&denom[dst * HEADS + h], ex);
}

__global__ void edge_alpha_kernel(const int* __restrict__ ei, float* __restrict__ score,
                                  const float* __restrict__ denom, int E, int NE) {
    int idx = blockIdx.x * blockDim.x + threadIdx.x;
    if (idx >= NE * HEADS) return;
    int e = idx / HEADS, h = idx - e * HEADS;
    int dst = (e < E) ? ei[E + e] : (e - E);
    score[idx] = score[idx] / (denom[dst * HEADS + h] + 1e-16f);
}

// One wave32 per edge; 6 x 32-lane float atomic-add scatter (L2-resident dst rows).
__global__ void aggregate_kernel(const int* __restrict__ ei, const float* __restrict__ xl,
                                 const float* __restrict__ alpha, float* __restrict__ out,
                                 int E, int NE) {
    int e = blockIdx.x * (blockDim.x >> 5) + (threadIdx.x >> 5);
    if (e >= NE) return;
    const int lane = threadIdx.x & 31;
    int src, dst;
    if (e < E) { src = ei[e]; dst = ei[E + e]; } else { src = dst = e - E; }
    const float* ps = xl + (size_t)src * DIM;
    float* po = out + (size_t)dst * DIM;
#pragma unroll
    for (int h = 0; h < HEADS; ++h) {
        float a = alpha[(size_t)e * HEADS + h];
        atomicAdd(&po[h * CDIM + lane], a * ps[h * CDIM + lane]);
    }
}

__global__ void finalize_kernel(const float* __restrict__ x, const float* __restrict__ bias,
                                float* __restrict__ out, int n) {
    int i = blockIdx.x * blockDim.x + threadIdx.x;
    if (i < n) {
        float v = x[i] + out[i] + bias[i % DIM];
        out[i] = v > 0.0f ? v : 0.0f;
    }
}

extern "C" void kernel_launch(void* const* d_in, const int* in_sizes, int n_in,
                              void* d_out, int out_size, void* d_ws, size_t ws_size,
                              hipStream_t stream) {
    const float* x    = (const float*)d_in[0];
    const float* Wl   = (const float*)d_in[1];
    const float* bl   = (const float*)d_in[2];
    const float* Wr   = (const float*)d_in[3];
    const float* br   = (const float*)d_in[4];
    const float* att  = (const float*)d_in[5];
    const float* bias = (const float*)d_in[6];
    const int*   ei   = (const int*)d_in[7];
    float* out = (float*)d_out;

    const int N  = in_sizes[0] / DIM;
    const int E  = in_sizes[7] / 2;
    const int NE = E + N;                       // edges + self loops

    // workspace carve-out (256B aligned)
    char* wsb = (char*)d_ws; size_t off = 0;
    auto alloc = [&](size_t bytes) -> void* {
        void* p = wsb + off;
        off += (bytes + 255) & ~(size_t)255;
        return p;
    };
    __bf16* xb    = (__bf16*)alloc((size_t)N * DIM * sizeof(__bf16));
    __bf16* Wb    = (__bf16*)alloc((size_t)2 * DIM * DIM * sizeof(__bf16));
    float*  xl    = (float*)alloc((size_t)N * DIM * sizeof(float));
    float*  xr    = (float*)alloc((size_t)N * DIM * sizeof(float));
    float*  score = (float*)alloc((size_t)NE * HEADS * sizeof(float));
    int*    smax  = (int*)alloc((size_t)N * HEADS * sizeof(int));
    float*  denom = (float*)alloc((size_t)N * HEADS * sizeof(float));

    const int nElem = N * DIM;
    cvt_x_kernel<<<(nElem + 255) / 256, 256, 0, stream>>>(x, xb, nElem);
    cvt_w_kernel<<<(2 * DIM * DIM + 255) / 256, 256, 0, stream>>>(Wl, Wr, Wb);

    dim3 gg((N + 15) / 16, (2 * DIM) / 64);     // 3125 x 6 blocks, 4 waves/block = 24 col tiles
    gemm_wmma_kernel<<<gg, 128, 0, stream>>>(xb, Wb, bl, br, xl, xr, N);

    init_kernel<<<(nElem + 255) / 256, 256, 0, stream>>>(out, smax, denom, nElem, N * HEADS);

    const int wavesPerBlock = 8;                // 256 threads = 8 wave32
    const int eb = (NE + wavesPerBlock - 1) / wavesPerBlock;
    edge_score_kernel<<<eb, 256, 0, stream>>>(ei, xl, xr, att, score, smax, E, NE);

    const int neh = NE * HEADS;
    edge_exp_kernel<<<(neh + 255) / 256, 256, 0, stream>>>(ei, score, smax, denom, E, NE);
    edge_alpha_kernel<<<(neh + 255) / 256, 256, 0, stream>>>(ei, score, denom, E, NE);

    aggregate_kernel<<<eb, 256, 0, stream>>>(ei, xl, score, out, E, NE);
    finalize_kernel<<<(nElem + 255) / 256, 256, 0, stream>>>(x, bias, out, nElem);
}
